// DeepseekV2Attention_20014547599363
// MI455X (gfx1250) — compile-verified, matrix operand order checked by reading, not run on hardware
//
#include <hip/hip_runtime.h>
#include <hip/hip_bf16.h>
#include <math.h>

// ---------------- problem constants (from reference) ----------------
#define S_LEN  2048
#define HDIM   4096
#define NHEADS 32
#define DN_    128
#define DR_    64
#define DV_    128
#define QK_    192      // DN + DR
#define RQ_    1536
#define RKV_   512
#define CKVP_  640      // RKV + DR (=576) padded to multiple of 128
#define EPS_   1e-6f

#define LDS_STRIDE 40   // halves per LDS tile row (32 data + 8 pad = 80 B)

typedef __attribute__((ext_vector_type(16))) _Float16 v16h;
typedef __attribute__((ext_vector_type(8)))  _Float16 v8h;
typedef __attribute__((ext_vector_type(8)))  float    v8f;

// ---------------- fp32 -> fp16 cast (grid-stride) ----------------
__global__ void __launch_bounds__(256)
cast_f32_to_f16(const float* __restrict__ in, _Float16* __restrict__ out, size_t n)
{
    size_t i = (size_t)blockIdx.x * blockDim.x + threadIdx.x;
    size_t stride = (size_t)gridDim.x * blockDim.x;
    for (; i < n; i += stride) out[i] = (_Float16)in[i];
}

__global__ void __launch_bounds__(256)
fill_zero_f16(_Float16* __restrict__ out, size_t n)
{
    size_t i = (size_t)blockIdx.x * blockDim.x + threadIdx.x;
    size_t stride = (size_t)gridDim.x * blockDim.x;
    for (; i < n; i += stride) out[i] = (_Float16)0.f;
}

// ---------------- CDNA5 async global->LDS copy (16 B per lane) ------
// GLOBAL_LOAD_ASYNC_TO_LDS_B128: vdst = LDS byte address VGPR,
// vaddr = 64-bit global address. Tracked by ASYNCcnt.
__device__ __forceinline__ void async_copy_b128(const void* gptr, void* lptr)
{
    unsigned lds = (unsigned)(size_t)lptr;            // low 32 bits = LDS offset
    unsigned long long g = (unsigned long long)(size_t)gptr;
    asm volatile("global_load_async_to_lds_b128 %0, %1, off"
                 :: "v"(lds), "v"(g) : "memory");
}
__device__ __forceinline__ void wait_async_0()
{
    asm volatile("s_wait_asynccnt 0x0" ::: "memory");
}
__device__ __forceinline__ void wait_async_4()
{
    asm volatile("s_wait_asynccnt 0x4" ::: "memory");
}

// stage a 128x32 f16 tile (rows row_base..+127, cols k0..+31) into LDS.
// 512 16-byte chunks, 256 threads -> 2 async copies per thread (2 per wave).
__device__ __forceinline__ void stage_tile_async(const _Float16* __restrict__ src,
                                                 int ld, int row_base, int k0,
                                                 _Float16* __restrict__ dst, int t)
{
#pragma unroll
    for (int i = 0; i < 2; ++i) {
        const int c   = t + i * 256;
        const int row = c >> 2;
        const int cc  = c & 3;
        const _Float16* g = src + (size_t)(row_base + row) * ld + k0 + cc * 8;
        _Float16* l = dst + row * LDS_STRIDE + cc * 8;
        async_copy_b128(g, l);
    }
}

// ---------------- WMMA fragment load from an LDS tile ----------------
// CDNA5 16-bit A-matrix 16x32 layout (ISA 7.12.2):
//   lanes 0-15 : row M=lane,    elems 0..7 = K 0..7,  elems 8..15 = K 16..23
//   lanes 16-31: row M=lane-16, elems 0..7 = K 8..15, elems 8..15 = K 24..31
// B operand (32x16, lane holds a column of B = a row of the (N,K) matrix)
// uses the identical per-lane pattern.
__device__ __forceinline__ v16h lds_frag(const _Float16* __restrict__ tile,
                                         int row0, int lane)
{
    const int half = lane >> 4;
    const _Float16* p = tile + (row0 + (lane & 15)) * LDS_STRIDE + 8 * half;
    v8h lo = *(const v8h*)(p);
    v8h hi = *(const v8h*)(p + 16);
    v16h f;
#pragma unroll
    for (int i = 0; i < 8; ++i) { f[i] = lo[i]; f[8 + i] = hi[i]; }
    return f;
}

// ---------------- tiled WMMA GEMM: C[m,n] = sum_k A[m,k] * B[n,k] ----
// A: M x K (row-major, lda), B: N x K (row-major, ldb = weight layout),
// C: M x N (row-major f32, ldc). Requires M%128==0, N%128==0 (or N==128),
// K%32==0, lda/ldb multiples of 8. Block: 256 threads = 8 waves;
// wave tile 32x64 (2x4 WMMA); block tile 128x128; double-buffered LDS
// staging via async global->LDS copies.
__global__ void __launch_bounds__(256)
gemm_f16f32(const _Float16* __restrict__ A, const _Float16* __restrict__ B,
            float* __restrict__ C, int M, int N, int K,
            int lda, int ldb, int ldc)
{
    __shared__ __align__(16) _Float16 smem[2][2][128 * LDS_STRIDE]; // 40 KB

    const int t    = threadIdx.x;
    const int lane = t & 31;
    const int wave = __builtin_amdgcn_readfirstlane(t >> 5);  // scalar wave id
    const int wr   = wave >> 1;                 // 0..3
    const int wc   = wave & 1;                  // 0..1
    const int m0   = blockIdx.y * 128;
    const int n0   = blockIdx.x * 128;

    v8f acc[2][4] = {};

    const int nk = K >> 5;   // K/32 tiles

    // prologue: stage tile 0 into buffer 0 (4 async instr per wave)
    stage_tile_async(A, lda, m0, 0, &smem[0][0][0], t);
    stage_tile_async(B, ldb, n0, 0, &smem[0][1][0], t);

    for (int kt = 0; kt < nk; ++kt) {
        const int buf = kt & 1;
        if (kt + 1 < nk) {
            // stage next tile into other buffer; previous iteration's trailing
            // barrier guarantees every wave is done reading it.
            stage_tile_async(A, lda, m0, (kt + 1) << 5, &smem[buf ^ 1][0][0], t);
            stage_tile_async(B, ldb, n0, (kt + 1) << 5, &smem[buf ^ 1][1][0], t);
            wait_async_4();   // retire current tile's 4 copies (in-order)
        } else {
            wait_async_0();
        }
        __syncthreads();      // all waves' copies for `buf` are now visible

        const _Float16* At = &smem[buf][0][0];
        const _Float16* Bt = &smem[buf][1][0];

        v16h a[2];
#pragma unroll
        for (int i = 0; i < 2; ++i)
            a[i] = lds_frag(At, wr * 32 + 16 * i, lane);
        v16h b[4];
#pragma unroll
        for (int j = 0; j < 4; ++j)
            b[j] = lds_frag(Bt, wc * 64 + 16 * j, lane);

#pragma unroll
        for (int i = 0; i < 2; ++i)
#pragma unroll
            for (int j = 0; j < 4; ++j)
                acc[i][j] = __builtin_amdgcn_wmma_f32_16x16x32_f16(
                    false, a[i], false, b[j], (short)0, acc[i][j], false, false);

        __syncthreads();      // protect buffers before they are refilled
    }

    // C/D layout: VGPR r -> (M = r + 8*(lane>=16), N = lane&15)
#pragma unroll
    for (int i = 0; i < 2; ++i) {
        const int mbase = m0 + wr * 32 + 16 * i + 8 * (lane >> 4);
#pragma unroll
        for (int j = 0; j < 4; ++j) {
            const int ncol = n0 + wc * 64 + 16 * j + (lane & 15);
#pragma unroll
            for (int r = 0; r < 8; ++r)
                C[(size_t)(mbase + r) * ldc + ncol] = acc[i][j][r];
        }
    }
}

// ---------------- RMSNorm (fp32 in, fp16 out), one block per row ----
__global__ void __launch_bounds__(256)
rmsnorm_f16(const float* __restrict__ x, const float* __restrict__ w,
            _Float16* __restrict__ out, int n, int in_ld, int out_ld)
{
    __shared__ float red[256];
    const int row = blockIdx.x;
    const float* xr = x + (size_t)row * in_ld;

    float ss = 0.f;
    for (int i = threadIdx.x; i < n; i += 256) { float v = xr[i]; ss += v * v; }
    red[threadIdx.x] = ss;
    __syncthreads();
    for (int s = 128; s > 0; s >>= 1) {
        if (threadIdx.x < s) red[threadIdx.x] += red[threadIdx.x + s];
        __syncthreads();
    }
    const float inv = rsqrtf(red[0] / (float)n + EPS_);
    for (int i = threadIdx.x; i < n; i += 256)
        out[(size_t)row * out_ld + i] = (_Float16)(xr[i] * inv * w[i]);
}

// ---------------- build Q/K (with RoPE) and V^T in fp16 -------------
// Q16/K16: [NH][S][QK] row-major; Vt16: [NH][DV][S]
__global__ void __launch_bounds__(128)
build_qkv(const float* __restrict__ q_f32,   // S x (NH*QK)
          const float* __restrict__ kv_f32,  // S x (NH*(DN+DV))
          const float* __restrict__ ckv_f32, // S x CKVP (640, cols 512..575 = k_rot)
          const float* __restrict__ freqs,   // S x (DR/2)
          _Float16* __restrict__ Q16, _Float16* __restrict__ K16,
          _Float16* __restrict__ Vt16)
{
    const int s = blockIdx.x;
    const int h = blockIdx.y;
    const int t = threadIdx.x;        // 0..127

    const float* qrow  = q_f32  + (size_t)s * (NHEADS * QK_) + h * QK_;
    const float* kvrow = kv_f32 + (size_t)s * (NHEADS * (DN_ + DV_)) + h * (DN_ + DV_);
    _Float16* Qr = Q16 + ((size_t)h * S_LEN + s) * QK_;
    _Float16* Kr = K16 + ((size_t)h * S_LEN + s) * QK_;

    // pass-through halves + transposed V
    Qr[t] = (_Float16)qrow[t];
    Kr[t] = (_Float16)kvrow[t];
    Vt16[((size_t)h * DV_ + t) * S_LEN + s] = (_Float16)kvrow[DN_ + t];

    if (t < DR_ / 2) {
        const float f = freqs[(size_t)s * (DR_ / 2) + t];
        const float c = cosf(f), sn = sinf(f);

        const float qxr = qrow[DN_ + 2 * t], qxi = qrow[DN_ + 2 * t + 1];
        Qr[DN_ + 2 * t]     = (_Float16)(qxr * c - qxi * sn);
        Qr[DN_ + 2 * t + 1] = (_Float16)(qxr * sn + qxi * c);

        const float kxr = ckv_f32[(size_t)s * CKVP_ + RKV_ + 2 * t];
        const float kxi = ckv_f32[(size_t)s * CKVP_ + RKV_ + 2 * t + 1];
        Kr[DN_ + 2 * t]     = (_Float16)(kxr * c - kxi * sn);
        Kr[DN_ + 2 * t + 1] = (_Float16)(kxr * sn + kxi * c);
    }
}

// ---------------- causal softmax: fp32 scores -> fp16 probs --------
__global__ void __launch_bounds__(256)
softmax_causal(const float* __restrict__ scores, _Float16* __restrict__ P, float scale)
{
    __shared__ float red[256];
    const int q = blockIdx.x;
    const int nvalid = q + 1;
    const float* srow = scores + (size_t)q * S_LEN;
    _Float16* prow = P + (size_t)q * S_LEN;

    float mx = -3.4e38f;
    for (int k = threadIdx.x; k < nvalid; k += 256) mx = fmaxf(mx, srow[k] * scale);
    red[threadIdx.x] = mx;
    __syncthreads();
    for (int s = 128; s > 0; s >>= 1) {
        if (threadIdx.x < s) red[threadIdx.x] = fmaxf(red[threadIdx.x], red[threadIdx.x + s]);
        __syncthreads();
    }
    mx = red[0];
    __syncthreads();

    float sum = 0.f;
    for (int k = threadIdx.x; k < nvalid; k += 256) sum += __expf(srow[k] * scale - mx);
    red[threadIdx.x] = sum;
    __syncthreads();
    for (int s = 128; s > 0; s >>= 1) {
        if (threadIdx.x < s) red[threadIdx.x] += red[threadIdx.x + s];
        __syncthreads();
    }
    const float inv = 1.f / red[0];

    for (int k = threadIdx.x; k < S_LEN; k += 256)
        prow[k] = (k < nvalid) ? (_Float16)(__expf(srow[k] * scale - mx) * inv)
                               : (_Float16)0.f;
}

// ------------------------------------------------------------------
extern "C" void kernel_launch(void* const* d_in, const int* in_sizes, int n_in,
                              void* d_out, int out_size, void* d_ws, size_t ws_size,
                              hipStream_t stream)
{
    const float* hidden  = (const float*)d_in[0];
    const float* freqs   = (const float*)d_in[1];
    const float* q_a_w   = (const float*)d_in[2];
    const float* q_a_ln  = (const float*)d_in[3];
    const float* q_b_w   = (const float*)d_in[4];
    const float* kv_a_w  = (const float*)d_in[5];
    const float* kv_a_ln = (const float*)d_in[6];
    const float* kv_b_w  = (const float*)d_in[7];
    const float* o_w     = (const float*)d_in[8];
    float* out = (float*)d_out;

    char* ws = (char*)d_ws;
    size_t off = 0;
    auto alloc = [&](size_t bytes) -> void* {
        void* p = ws + off;
        off = (off + bytes + 255) & ~(size_t)255;
        return p;
    };

    // fp16 operand copies (kv_a padded from 576 to 640 rows)
    _Float16* h16       = (_Float16*)alloc((size_t)S_LEN * HDIM * 2);
    _Float16* q_a_w16   = (_Float16*)alloc((size_t)RQ_ * HDIM * 2);
    _Float16* q_b_w16   = (_Float16*)alloc((size_t)NHEADS * QK_ * RQ_ * 2);
    _Float16* kv_a_w16  = (_Float16*)alloc((size_t)CKVP_ * HDIM * 2);
    _Float16* kv_b_w16  = (_Float16*)alloc((size_t)NHEADS * (DN_ + DV_) * RKV_ * 2);
    _Float16* o_w16     = (_Float16*)alloc((size_t)HDIM * NHEADS * DV_ * 2);
    // activations
    float*    q_a_f32   = (float*)alloc((size_t)S_LEN * RQ_ * 4);
    _Float16* q_a_n16   = (_Float16*)alloc((size_t)S_LEN * RQ_ * 2);
    float*    q_f32     = (float*)alloc((size_t)S_LEN * NHEADS * QK_ * 4);
    float*    ckv_f32   = (float*)alloc((size_t)S_LEN * CKVP_ * 4);
    _Float16* ckv_n16   = (_Float16*)alloc((size_t)S_LEN * RKV_ * 2);
    float*    kv_f32    = (float*)alloc((size_t)S_LEN * NHEADS * (DN_ + DV_) * 4);
    _Float16* Q16       = (_Float16*)alloc((size_t)NHEADS * S_LEN * QK_ * 2);
    _Float16* K16       = (_Float16*)alloc((size_t)NHEADS * S_LEN * QK_ * 2);
    _Float16* Vt16      = (_Float16*)alloc((size_t)NHEADS * DV_ * S_LEN * 2);
    float*    scores    = (float*)alloc((size_t)S_LEN * S_LEN * 4);     // per-head reuse (L2-resident)
    _Float16* P16       = (_Float16*)alloc((size_t)S_LEN * S_LEN * 2);  // per-head reuse
    float*    O_f32     = (float*)alloc((size_t)S_LEN * NHEADS * DV_ * 4);
    _Float16* O16       = (_Float16*)alloc((size_t)S_LEN * NHEADS * DV_ * 2);
    (void)ws_size; (void)in_sizes; (void)n_in; (void)out_size;

    auto cast16 = [&](const float* in, _Float16* o, size_t n) {
        size_t nb = (n + 255) / 256; if (nb > 4096) nb = 4096;
        cast_f32_to_f16<<<dim3((unsigned)nb), 256, 0, stream>>>(in, o, n);
    };
    auto gemm = [&](const _Float16* A, const _Float16* B, float* C,
                    int M, int N, int K, int lda, int ldb, int ldc) {
        dim3 grid((N + 127) / 128, (M + 127) / 128);
        gemm_f16f32<<<grid, 256, 0, stream>>>(A, B, C, M, N, K, lda, ldb, ldc);
    };

    // --- operand conversion ---
    cast16(hidden, h16,      (size_t)S_LEN * HDIM);
    cast16(q_a_w,  q_a_w16,  (size_t)RQ_ * HDIM);
    cast16(q_b_w,  q_b_w16,  (size_t)NHEADS * QK_ * RQ_);
    cast16(kv_a_w, kv_a_w16, (size_t)(RKV_ + DR_) * HDIM);
    // zero the 64 padded weight rows (576..639)
    fill_zero_f16<<<256, 256, 0, stream>>>(kv_a_w16 + (size_t)(RKV_ + DR_) * HDIM,
                                           (size_t)(CKVP_ - RKV_ - DR_) * HDIM);
    cast16(kv_b_w, kv_b_w16, (size_t)NHEADS * (DN_ + DV_) * RKV_);
    cast16(o_w,    o_w16,    (size_t)HDIM * NHEADS * DV_);

    // --- Q path: q_a GEMM -> RMSNorm -> q_b GEMM ---
    gemm(h16, q_a_w16, q_a_f32, S_LEN, RQ_, HDIM, HDIM, HDIM, RQ_);
    rmsnorm_f16<<<S_LEN, 256, 0, stream>>>(q_a_f32, q_a_ln, q_a_n16, RQ_, RQ_, RQ_);
    gemm(q_a_n16, q_b_w16, q_f32, S_LEN, NHEADS * QK_, RQ_, RQ_, RQ_, NHEADS * QK_);

    // --- KV path: kv_a GEMM (N padded to 640) -> RMSNorm(512) -> kv_b GEMM ---
    gemm(h16, kv_a_w16, ckv_f32, S_LEN, CKVP_, HDIM, HDIM, HDIM, CKVP_);
    rmsnorm_f16<<<S_LEN, 256, 0, stream>>>(ckv_f32, kv_a_ln, ckv_n16, RKV_, CKVP_, RKV_);
    gemm(ckv_n16, kv_b_w16, kv_f32, S_LEN, NHEADS * (DN_ + DV_), RKV_,
         RKV_, RKV_, NHEADS * (DN_ + DV_));

    // --- RoPE + head layout (Q, K per-head, V transposed) ---
    build_qkv<<<dim3(S_LEN, NHEADS), 128, 0, stream>>>(q_f32, kv_f32, ckv_f32,
                                                       freqs, Q16, K16, Vt16);

    // --- attention per head: QK^T (WMMA) -> causal softmax -> PV (WMMA) ---
    const float scale = 1.0f / sqrtf((float)QK_);
    for (int h = 0; h < NHEADS; ++h) {
        const _Float16* Qh  = Q16 + (size_t)h * S_LEN * QK_;
        const _Float16* Kh  = K16 + (size_t)h * S_LEN * QK_;
        const _Float16* Vth = Vt16 + (size_t)h * DV_ * S_LEN;

        gemm(Qh, Kh, scores, S_LEN, S_LEN, QK_, QK_, QK_, S_LEN);
        softmax_causal<<<S_LEN, 256, 0, stream>>>(scores, P16, scale);
        gemm(P16, Vth, O_f32 + (size_t)h * DV_, S_LEN, DV_, S_LEN,
             S_LEN, S_LEN, NHEADS * DV_);
    }

    // --- output projection ---
    cast16(O_f32, O16, (size_t)S_LEN * NHEADS * DV_);
    gemm(O16, o_w16, out, S_LEN, HDIM, NHEADS * DV_, NHEADS * DV_, NHEADS * DV_, HDIM);
}
